// SingleHeadedAttention_5798205849845
// MI455X (gfx1250) — matrix operand order repaired and checked by reading, not compiled, not run on hardware
//
#include <hip/hip_runtime.h>
#include <math.h>
#include <stdint.h>

// B=2, LQ=LKV=256, D=512. Output: softmax weights [B, LQ, LKV] float32.
// v / W_v / b_v are computed-but-unused in the reference -> skipped.

#define BATCH   2
#define LQ      256
#define LKV     256
#define DIM     512
#define MROWS   (BATCH * LQ)      // 512 (batch folds into GEMM M dim)

typedef _Float16 v16h __attribute__((ext_vector_type(16)));
typedef float    v8f  __attribute__((ext_vector_type(8)));

__device__ __forceinline__ float fast_tanh(float x) {
#if __has_builtin(__builtin_amdgcn_tanhf)
    return __builtin_amdgcn_tanhf(x);          // v_tanh_f32 (CDNA5 TRANS op)
#elif __has_builtin(__builtin_amdgcn_tanh_f32)
    return __builtin_amdgcn_tanh_f32(x);
#else
    return tanhf(x);
#endif
}

// ---------------------------------------------------------------------------
// WMMA GEMM:  Out[m,n] = (sum_e A[m,e] * W[n,e] + bias[n]) * scale
// 128-thread block computes a 64x64 tile; wave w owns a 16x64 strip, so the
// A fragment is gathered once per K-step and reused by 4 v_wmma ops.
// Fragment gathers follow ISA 7.12.2 wave32 layouts.
// ---------------------------------------------------------------------------
template <bool A_IS_F16, bool OUT_F16>
__global__ __launch_bounds__(128)
void wmma_gemm_nt(const void* __restrict__ Aptr, int lda,
                  const float* __restrict__ W, int ldw,
                  const float* __restrict__ bias, float scale,
                  void* __restrict__ Out, int ldo, int E)
{
    __shared__ _Float16 aT[64][34];   // A tile: [m][k], padded
    __shared__ _Float16 wT[64][34];   // W tile: [n][k], padded

    const int tid  = threadIdx.x;
    const int lane = tid & 31;
    const int wave = tid >> 5;        // 0..3
    const int m0 = blockIdx.y * 64;
    const int n0 = blockIdx.x * 64;

    v8f acc[4] = {};

    for (int e0 = 0; e0 < E; e0 += 32) {
        // Stage: wave w loads rows w*16..w*16+15 (lane = k column, coalesced).
        #pragma unroll
        for (int i = 0; i < 16; ++i) {
            const int row = wave * 16 + i;
            float av;
            if (A_IS_F16)
                av = (float)((const _Float16*)Aptr)[(size_t)(m0 + row) * lda + e0 + lane];
            else
                av = ((const float*)Aptr)[(size_t)(m0 + row) * lda + e0 + lane];
            aT[row][lane] = (_Float16)av;
            wT[row][lane] = (_Float16)W[(size_t)(n0 + row) * ldw + e0 + lane];
        }
        __syncthreads();

        const int hi = lane >> 4;     // lane half
        const int rr = lane & 15;     // row (A) / col (B) index

        // A fragment (16-bit A 16x32, ISA 7.12.2):
        //  low lanes : V0..3 -> K0..7,  V4..7 -> K16..23
        //  high lanes: V0..3 -> K8..15, V4..7 -> K24..31
        v16h afrag;
        #pragma unroll
        for (int i = 0; i < 16; ++i) {
            const int j = i >> 1, r = i & 1;
            const int k = ((j & 4) ? 16 : 0) + hi * 8 + ((j & 3) << 1) + r;
            afrag[i] = aT[wave * 16 + rr][k];
        }

        // 4 N-subtiles reuse the same A fragment.
        #pragma unroll
        for (int nt = 0; nt < 4; ++nt) {
            v16h bfrag;
            #pragma unroll
            for (int i = 0; i < 16; ++i)      // lane = column n, K = (i>>3)*16 + hi*8 + (i&7)
                bfrag[i] = wT[nt * 16 + rr][((i >> 3) << 4) + hi * 8 + (i & 7)];
            acc[nt] = __builtin_amdgcn_wmma_f32_16x16x32_f16(
                false, afrag, false, bfrag, (short)0, acc[nt], false, false);
        }
        __syncthreads();
    }

    // C/D layout: lane -> N = lane&15 ; VGPR i -> M = (lane>>4)*8 + i.
    const int nl  = lane & 15;
    const int hi2 = lane >> 4;
    #pragma unroll
    for (int nt = 0; nt < 4; ++nt) {
        const int n = n0 + nt * 16 + nl;
        const float bv = bias ? bias[n] : 0.0f;
        #pragma unroll
        for (int i = 0; i < 8; ++i) {
            const int m = m0 + wave * 16 + hi2 * 8 + i;
            const float v = (acc[nt][i] + bv) * scale;
            if (OUT_F16)
                ((_Float16*)Out)[(size_t)m * ldo + n] = (_Float16)v;
            else
                ((float*)Out)[(size_t)m * ldo + n] = v;
        }
    }
}

// ---------------------------------------------------------------------------
// Logits + mask + softmax. One block (256 threads = 8 waves) per (b, q) row.
// logit[k] = b_l + sum_d tanh(AQ[bq,d] + AK[bk,d]) * W_l[d]   (b_c folded
// into AK by the GEMM epilogue). Dominant cost: 512 v_tanh_f32 per (q,k).
//
// Each wave double-buffers its AK row into LDS with the CDNA5 async engine:
// 4x GLOBAL_LOAD_ASYNC_TO_LDS_B128 (32 lanes x 16B = 512B each) per 2KB row,
// pipelined one row ahead; ASYNCcnt completes in-order, so s_wait_asynccnt 4
// after issuing the next row guarantees the current buffer has landed.
// ---------------------------------------------------------------------------
__global__ __launch_bounds__(256)
void attn_logits_softmax(const float* __restrict__ AQ,
                         const float* __restrict__ AK,
                         const float* __restrict__ Wl,
                         const float* __restrict__ blp,
                         const int*   __restrict__ mask,
                         float* __restrict__ out)
{
    __shared__ __align__(16) float akbuf[2][8][DIM];   // 32 KB double buffer
    __shared__ __align__(16) float aq[DIM];
    __shared__ __align__(16) float wl[DIM];
    __shared__ float logits[LKV];
    __shared__ float red[8];
    __shared__ int   anyUn;

    const int t  = threadIdx.x;
    const int bq = blockIdx.x;          // 0..511
    const int b  = bq >> 8;

    aq[t]       = AQ[(size_t)bq * DIM + t];
    aq[t + 256] = AQ[(size_t)bq * DIM + t + 256];
    wl[t]       = Wl[t];
    wl[t + 256] = Wl[t + 256];
    if (t == 0) anyUn = 0;
    __syncthreads();

    const int wave = t >> 5;
    const int lane = t & 31;

    const float*   akbase = AK + (size_t)b * LKV * DIM;
    const uint32_t goff   = (uint32_t)(lane * 16);     // per-lane byte offset

    // Issue one 2KB AK row (4 x b128 per lane-set) into LDS buffer `buf`.
    auto issue_row = [&](int row, int buf) {
        const uint64_t sbase = (uint64_t)(uintptr_t)(akbase + (size_t)row * DIM);
        const uint32_t ldso  =
            (uint32_t)(uintptr_t)(&akbuf[buf][wave][0]) + goff;
        // INST_OFFSET is added to both the LDS and global addresses (ISA 10.2),
        // so the four row-chunks differ only in the immediate.
        asm volatile(
            "global_load_async_to_lds_b128 %0, %1, %2 offset:0\n\t"
            "global_load_async_to_lds_b128 %0, %1, %2 offset:512\n\t"
            "global_load_async_to_lds_b128 %0, %1, %2 offset:1024\n\t"
            "global_load_async_to_lds_b128 %0, %1, %2 offset:1536"
            :: "v"(ldso), "v"(goff), "s"(sbase) : "memory");
    };

    issue_row(wave, 0);                  // prime the pipeline

    for (int it = 0; it < LKV / 8; ++it) {
        const int  k0   = wave + it * 8;
        const bool more = (it + 1) < (LKV / 8);
        if (more) issue_row(k0 + 8, (it + 1) & 1);

        if (more) asm volatile("s_wait_asynccnt 0x4" ::: "memory");
        else      asm volatile("s_wait_asynccnt 0x0" ::: "memory");

        const float4* ak4 = (const float4*)&akbuf[it & 1][wave][0];
        const float4* aq4 = (const float4*)aq;
        const float4* wl4 = (const float4*)wl;

        float s = 0.0f;
        #pragma unroll
        for (int i = 0; i < 4; ++i) {           // ds_load_b128 x3 per step
            const float4 a = aq4[lane + i * 32];
            const float4 k = ak4[lane + i * 32];
            const float4 w = wl4[lane + i * 32];
            s += fast_tanh(a.x + k.x) * w.x;
            s += fast_tanh(a.y + k.y) * w.y;
            s += fast_tanh(a.z + k.z) * w.z;
            s += fast_tanh(a.w + k.w) * w.w;
        }
        #pragma unroll
        for (int off = 16; off; off >>= 1)
            s += __shfl_xor(s, off);
        if (lane == 0) logits[k0] = s;
    }

    const int mk = (mask[(size_t)bq * LKV + t] != 0);
    if (mk) atomicOr(&anyUn, 1);
    __syncthreads();

    float lg = logits[t] + blp[0];
    if (anyUn && !mk) lg = -INFINITY;   // fully-masked rows stay unmasked

    // block max (wave shuffle + cross-wave LDS)
    float m = lg;
    #pragma unroll
    for (int off = 16; off; off >>= 1) m = fmaxf(m, __shfl_xor(m, off));
    if (lane == 0) red[wave] = m;
    __syncthreads();
    float mm = red[0];
    #pragma unroll
    for (int i = 1; i < 8; ++i) mm = fmaxf(mm, red[i]);

    const float e = __expf(lg - mm);    // v_exp_f32; exp(-inf - mm) == 0

    float s = e;
    #pragma unroll
    for (int off = 16; off; off >>= 1) s += __shfl_xor(s, off);
    __syncthreads();                    // everyone done reading red (max phase)
    if (lane == 0) red[wave] = s;
    __syncthreads();
    float tot = red[0];
    #pragma unroll
    for (int i = 1; i < 8; ++i) tot += red[i];

    out[(size_t)bq * LKV + t] = e / tot;
}

// ---------------------------------------------------------------------------
// Input order: key, query, value, mask, W_k, b_k, W_v, b_v, W_q, b_q,
//              W_c, b_c, W_l, b_l
// ---------------------------------------------------------------------------
extern "C" void kernel_launch(void* const* d_in, const int* in_sizes, int n_in,
                              void* d_out, int out_size, void* d_ws, size_t ws_size,
                              hipStream_t stream)
{
    const float* key   = (const float*)d_in[0];
    const float* query = (const float*)d_in[1];
    const int*   mask  = (const int*)  d_in[3];
    const float* W_k   = (const float*)d_in[4];
    const float* b_k   = (const float*)d_in[5];
    const float* W_q   = (const float*)d_in[8];
    const float* b_q   = (const float*)d_in[9];
    const float* W_c   = (const float*)d_in[10];
    const float* b_c   = (const float*)d_in[11];
    const float* W_l   = (const float*)d_in[12];
    const float* b_l   = (const float*)d_in[13];

    // workspace layout (3 MB total)
    _Float16* Qh = (_Float16*)d_ws;                     // [512,512] f16
    _Float16* Kh = Qh + (size_t)MROWS * DIM;            // [512,512] f16
    float*    AQ = (float*)(Kh + (size_t)MROWS * DIM);  // [512,512] f32
    float*    AK = AQ + (size_t)MROWS * DIM;            // [512,512] f32

    const dim3 grid(DIM / 64, MROWS / 64);  // 8 x 8 blocks of 64x64 tiles
    const dim3 blk(128);
    const float inv_sqrt_d = 0.04419417382415922f;  // 1/sqrt(512)

    // Q = (query @ W_q.T + b_q) / sqrt(D)   -> f16
    wmma_gemm_nt<false, true><<<grid, blk, 0, stream>>>(
        query, DIM, W_q, DIM, b_q, inv_sqrt_d, Qh, DIM, DIM);
    // K = key @ W_k.T + b_k                 -> f16
    wmma_gemm_nt<false, true><<<grid, blk, 0, stream>>>(
        key, DIM, W_k, DIM, b_k, 1.0f, Kh, DIM, DIM);
    // AQ = Q @ Wc[:, :D].T                  -> f32
    wmma_gemm_nt<true, false><<<grid, blk, 0, stream>>>(
        Qh, DIM, W_c, 2 * DIM, nullptr, 1.0f, AQ, DIM, DIM);
    // AK = K @ Wc[:, D:].T + b_c            -> f32 (b_c folded in here)
    wmma_gemm_nt<true, false><<<grid, blk, 0, stream>>>(
        Kh, DIM, W_c + DIM, 2 * DIM, b_c, 1.0f, AK, DIM, DIM);

    attn_logits_softmax<<<BATCH * LQ, 256, 0, stream>>>(
        AQ, AK, W_l, b_l, mask, (float*)d_out);
}
